// NormalShieldedAttentionWithRPE_39745627357526
// MI455X (gfx1250) — matrix operand order repaired and checked by reading, not compiled, not run on hardware
//
#include <hip/hip_runtime.h>
#include <hip/hip_bf16.h>

// Problem constants (reference: B,S,H,D,DM = 2,512,8,32,256)
#define BB 2
#define SS 512
#define HH 8
#define DD 32
#define DMD 256
#define HD 256   // H*D
#define NC (SS / 32)   // 16 RPE chunks per (b,i)
#define RPAD 36        // LDS row stride (floats): 16B-aligned rows, only 2-way bank conflict

typedef __bf16 v16bf __attribute__((ext_vector_type(16)));
typedef __bf16 v8bf  __attribute__((ext_vector_type(8)));
typedef float  v8f   __attribute__((ext_vector_type(8)));
typedef int    v4i   __attribute__((ext_vector_type(4)));

#define AS1 __attribute__((address_space(1)))
#define AS3 __attribute__((address_space(3)))

#if defined(__has_builtin)
#if __has_builtin(__builtin_amdgcn_global_load_async_to_lds_b128) && \
    __has_builtin(__builtin_amdgcn_s_wait_asynccnt)
#define USE_ASYNC_LDS 1
#endif
#endif

// ---- WMMA bf16 fragment loaders: pure contiguous vector loads (b128) ----
// A tile 16(M)x32(K) from row-major bf16 (base = &A[row0][k0], ld in elements).
// Per ISA: lane(0..15)=row M, elems 0..7 -> K=8*half+0..7 ; elems 8..15 -> K=16+8*half+0..7
__device__ __forceinline__ v16bf load_a_rm(const __bf16* __restrict__ base, int ld, int lane) {
  const int m = lane & 15, off = (lane >> 4) * 8;
  const __bf16* p = base + m * ld + off;
  v8bf lo = *(const v8bf*)(p);
  v8bf hi = *(const v8bf*)(p + 16);
  return __builtin_shufflevector(lo, hi, 0, 1, 2, 3, 4, 5, 6, 7, 8, 9, 10, 11, 12, 13, 14, 15);
}

// B tile 32(K)x16(N) from N-major (transposed) bf16 (base = &BT[col0][k0], ldT in elements).
// Per ISA: lane n=lane%16, elems 0..7 -> K=8*half+0..7 ; elems 8..15 -> K=16+8*half+0..7
__device__ __forceinline__ v16bf load_b_nm(const __bf16* __restrict__ base, int ldT, int lane) {
  const int n = lane & 15, off = (lane >> 4) * 8;
  const __bf16* p = base + n * ldT + off;
  v8bf lo = *(const v8bf*)(p);
  v8bf hi = *(const v8bf*)(p + 16);
  return __builtin_shufflevector(lo, hi, 0, 1, 2, 3, 4, 5, 6, 7, 8, 9, 10, 11, 12, 13, 14, 15);
}

// ---- Kernel 0a: hidden fp32 -> bf16 row-major copy ----
__global__ __launch_bounds__(256)
void k_cvt_x(const float* __restrict__ x, __bf16* __restrict__ xb) {
  const int t = blockIdx.x * 256 + threadIdx.x;  // 0 .. 262143
  xb[t] = (__bf16)x[t];
}

// ---- Kernel 0b: transpose-convert {w_q,w_k,w_v,fc_w} (256x256) -> N-major bf16 ----
__global__ __launch_bounds__(256)
void k_cvt_w(const float* __restrict__ wq, const float* __restrict__ wk,
             const float* __restrict__ wv, const float* __restrict__ fcw,
             __bf16* __restrict__ WT) {
  const int z = blockIdx.z;
  const float* src = (z == 0) ? wq : (z == 1) ? wk : (z == 2) ? wv : fcw;
  const int t = blockIdx.x * 256 + threadIdx.x;  // 0 .. 65535
  const int n = t >> 8, k = t & 255;
  WT[(size_t)z * HD * DMD + t] = (__bf16)src[k * DMD + n];  // WT[n][k] = W[k][n]
}

// ---- Kernel 1: fused QKV projection; q,k fp32 in (b,h,s,d); v -> bf16 transposed (b,h,d,s)
__global__ __launch_bounds__(32)
void k_qkv(const __bf16* __restrict__ Xb, const __bf16* __restrict__ WT,
           float* __restrict__ qo, float* __restrict__ ko, __bf16* __restrict__ vT) {
  const int lane = threadIdx.x;
  const int ct = blockIdx.x, rt = blockIdx.y, mat = blockIdx.z;
  const __bf16* Wt = WT + (size_t)mat * HD * DMD + ct * 16 * DMD;
  v8f c = {};
#pragma unroll
  for (int k0 = 0; k0 < DMD; k0 += 32) {
    v16bf a = load_a_rm(Xb + rt * 16 * DMD + k0, DMD, lane);
    v16bf b = load_b_nm(Wt + k0, DMD, lane);
    c = __builtin_amdgcn_wmma_f32_16x16x32_bf16(false, a, false, b, (short)0, c, false, false);
  }
  const int half = lane >> 4, n = lane & 15;
  const int col = ct * 16 + n, h = col >> 5, d = col & 31;
  const int m0 = rt * 16 + 8 * half;     // first of 8 consecutive rows
  const int bi = m0 >> 9, s0 = m0 & 511;
  if (mat == 2) {                        // V: store transposed bf16, contiguous in s
    v8bf pack;
#pragma unroll
    for (int r = 0; r < 8; ++r) pack[r] = (__bf16)c[r];
    *(v8bf*)(vT + ((size_t)(bi * HH + h) * DD + d) * SS + s0) = pack;
  } else {
    float* O = (mat == 0) ? qo : ko;
#pragma unroll
    for (int r = 0; r < 8; ++r)
      O[(((bi * HH) + h) * SS + (s0 + r)) * DD + d] = c[r];
  }
}

// ---- Kernel 2: scores + mask + softmax fused; one block per (b,i); 8 waves = 8 heads.
// RPE chunks (32 j x 32 d) double-buffered through LDS via async-to-LDS (<=2 in flight),
// shared across all 8 heads so the 64MB RPE tensor is read exactly once.
__global__ __launch_bounds__(256)
void k_scores(const float* __restrict__ q, const float* __restrict__ kmat,
              const float* __restrict__ rpe, const int* __restrict__ station_p,
              __bf16* __restrict__ attnb) {
  __shared__ float rpeT[2][32 * RPAD];
  __shared__ float kT[HH * DD];
  const int tid = threadIdx.x;
  const int lane = tid & 31, h = tid >> 5;
  const int b = blockIdx.x >> 9, i = blockIdx.x & 511;
  const int station = station_p[0];

  kT[tid] = kmat[(((b * HH) + h) * SS + i) * DD + lane];

  const size_t rbase = ((size_t)(b * SS + i)) * SS * DD;  // chunk jc = 1024 contiguous floats
  const float* gsrc = rpe + rbase + tid * 4;              // this thread's 16B within a chunk
  float sc[NC];

#ifdef USE_ASYNC_LDS
  float* ld0 = &rpeT[0][(tid >> 3) * RPAD + (tid & 7) * 4];
  float* ld1 = &rpeT[1][(tid >> 3) * RPAD + (tid & 7) * 4];
  __builtin_amdgcn_global_load_async_to_lds_b128((AS1 v4i*)gsrc, (AS3 v4i*)ld0, 0, 0);
#endif

#pragma unroll 1
  for (int jc = 0; jc < NC; ++jc) {
    const int cur = jc & 1;
#ifdef USE_ASYNC_LDS
    if (jc + 1 < NC) {   // prefetch next chunk into the other buffer; then wait for current
      __builtin_amdgcn_global_load_async_to_lds_b128(
          (AS1 v4i*)(gsrc + (size_t)(jc + 1) * (32 * DD)), (AS3 v4i*)(cur ? ld0 : ld1), 0, 0);
      __builtin_amdgcn_s_wait_asynccnt(1);
    } else {
      __builtin_amdgcn_s_wait_asynccnt(0);
    }
#else
#pragma unroll
    for (int r = 0; r < 4; ++r) {
      const int idx = r * 256 + tid;
      rpeT[cur][(idx >> 5) * RPAD + (idx & 31)] = rpe[rbase + (size_t)jc * (32 * DD) + idx];
    }
    if (jc + 1 < NC)
      __builtin_prefetch(rpe + rbase + (size_t)(jc + 1) * (32 * DD) + tid * 4, 0, 1);
#endif
    __syncthreads();   // buf[cur] visible to all waves (covers kT on jc==0)

    const int j = jc * 32 + lane;
    const float* qrow = q + (((b * HH) + h) * SS + j) * DD;
    float acc = 0.f;
#pragma unroll
    for (int d = 0; d < DD; ++d)
      acc += qrow[d] * rpeT[cur][lane * RPAD + d] * kT[h * 32 + d];
    acc *= 0.17677669529663689f;                 // 1/sqrt(32)
    sc[jc] = (j < station || j == i) ? acc : -1e10f;

    __syncthreads();   // everyone done with buf[cur] before it is overwritten at jc+2
  }

  // softmax over 512 j's (16 per lane x 32 lanes), wave32 shuffle reductions
  float mx = sc[0];
#pragma unroll
  for (int t = 1; t < NC; ++t) mx = fmaxf(mx, sc[t]);
#pragma unroll
  for (int off = 16; off > 0; off >>= 1) mx = fmaxf(mx, __shfl_xor(mx, off, 32));
  float sum = 0.f;
#pragma unroll
  for (int t = 0; t < NC; ++t) { sc[t] = __expf(sc[t] - mx); sum += sc[t]; }
#pragma unroll
  for (int off = 16; off > 0; off >>= 1) sum += __shfl_xor(sum, off, 32);
  const float inv = 1.f / sum;

  __bf16* arow = attnb + ((size_t)((b * HH + h) * SS + i)) * SS;
#pragma unroll
  for (int t = 0; t < NC; ++t) arow[t * 32 + lane] = (__bf16)(sc[t] * inv);
}

// ---- Kernel 3: O = P @ V per (b,h); P bf16 512x512 row-major, V bf16 transposed (d-major).
// Pure b128 loads + WMMA. Output bf16 in (b, s, h*d) for the FC GEMM.
__global__ __launch_bounds__(32)
void k_pv(const __bf16* __restrict__ attnb, const __bf16* __restrict__ vT,
          __bf16* __restrict__ atto) {
  const int lane = threadIdx.x;
  const int nt = blockIdx.x, mt = blockIdx.y, bh = blockIdx.z;
  const __bf16* P  = attnb + (size_t)bh * SS * SS + mt * 16 * SS;
  const __bf16* Vt = vT   + (size_t)bh * DD * SS + nt * 16 * SS;
  v8f c = {};
#pragma unroll
  for (int k0 = 0; k0 < SS; k0 += 32) {
    v16bf a = load_a_rm(P + k0, SS, lane);
    v16bf b = load_b_nm(Vt + k0, SS, lane);
    c = __builtin_amdgcn_wmma_f32_16x16x32_bf16(false, a, false, b, (short)0, c, false, false);
  }
  const int half = lane >> 4, n = lane & 15;
  const int bb = bh >> 3, h = bh & 7;
  const int d = nt * 16 + n;
#pragma unroll
  for (int r = 0; r < 8; ++r) {
    const int i = mt * 16 + r + 8 * half;
    atto[((size_t)(bb * SS + i)) * HD + h * DD + d] = (__bf16)c[r];
  }
}

// ---- Kernel 4: ctx = A @ fc_w + fc_b + residual (WMMA, fused epilogue, fp32 out) ----
__global__ __launch_bounds__(32)
void k_fc(const __bf16* __restrict__ A, const __bf16* __restrict__ WTfc,
          const float* __restrict__ bias, const float* __restrict__ resid,
          float* __restrict__ ctx) {
  const int lane = threadIdx.x;
  const int ct = blockIdx.x, rt = blockIdx.y;
  v8f c = {};
#pragma unroll
  for (int k0 = 0; k0 < HD; k0 += 32) {
    v16bf a = load_a_rm(A + rt * 16 * HD + k0, HD, lane);
    v16bf b = load_b_nm(WTfc + ct * 16 * DMD + k0, DMD, lane);
    c = __builtin_amdgcn_wmma_f32_16x16x32_bf16(false, a, false, b, (short)0, c, false, false);
  }
  const int half = lane >> 4, n = lane & 15;
  const int col = ct * 16 + n;
#pragma unroll
  for (int r = 0; r < 8; ++r) {
    const int m = rt * 16 + r + 8 * half;
    ctx[m * DMD + col] = c[r] + bias[col] + resid[m * DMD + col];
  }
}

// ---- Kernel 5: LayerNorm, one wave per row (256 cols -> 8/lane) ----
__global__ __launch_bounds__(32)
void k_ln(const float* __restrict__ ctx, const float* __restrict__ w,
          const float* __restrict__ bsr, float* __restrict__ out) {
  const int lane = threadIdx.x;
  const int row = blockIdx.x;
  float x[DMD / 32];
  float s = 0.f;
#pragma unroll
  for (int t = 0; t < DMD / 32; ++t) { x[t] = ctx[row * DMD + t * 32 + lane]; s += x[t]; }
#pragma unroll
  for (int off = 16; off > 0; off >>= 1) s += __shfl_xor(s, off, 32);
  const float mu = s * (1.f / DMD);
  float vs = 0.f;
#pragma unroll
  for (int t = 0; t < DMD / 32; ++t) { const float d = x[t] - mu; vs += d * d; }
#pragma unroll
  for (int off = 16; off > 0; off >>= 1) vs += __shfl_xor(vs, off, 32);
  const float rstd = rsqrtf(vs * (1.f / DMD) + 1e-6f);
#pragma unroll
  for (int t = 0; t < DMD / 32; ++t) {
    const int c = t * 32 + lane;
    out[row * DMD + c] = (x[t] - mu) * rstd * w[c] + bsr[c];
  }
}

extern "C" void kernel_launch(void* const* d_in, const int* in_sizes, int n_in,
                              void* d_out, int out_size, void* d_ws, size_t ws_size,
                              hipStream_t stream) {
  const float* hidden = (const float*)d_in[0];
  const float* rpe    = (const float*)d_in[1];
  const float* w_q    = (const float*)d_in[2];
  const float* w_k    = (const float*)d_in[3];
  const float* w_v    = (const float*)d_in[4];
  const float* fc_w   = (const float*)d_in[5];
  const float* fc_b   = (const float*)d_in[6];
  const float* ln_w   = (const float*)d_in[7];
  const float* ln_b   = (const float*)d_in[8];
  const int*   station = (const int*)d_in[9];
  float* out = (float*)d_out;

  // workspace layout: fp32 q,k,ctx  then bf16 Xb, WT(4), vT, atto, attnb  (~13.6 MB)
  const size_t NQ = (size_t)BB * HH * SS * DD;       // 262144
  float* q   = (float*)d_ws;
  float* k   = q + NQ;
  float* ctx = k + NQ;
  __bf16* Xb    = (__bf16*)(ctx + NQ);
  __bf16* WT    = Xb + (size_t)BB * SS * DMD;        // 4 x 256x256 transposed bf16
  __bf16* vT    = WT + (size_t)4 * HD * DMD;         // (b,h,d,s) bf16
  __bf16* atto  = vT + NQ;                           // (b,s,h*d) bf16
  __bf16* attnb = atto + (size_t)BB * SS * HD;       // (b,h,i,j) bf16 probs

  k_cvt_x <<<dim3((BB * SS * DMD) / 256), 256, 0, stream>>>(hidden, Xb);
  k_cvt_w <<<dim3((HD * DMD) / 256, 1, 4), 256, 0, stream>>>(w_q, w_k, w_v, fc_w, WT);
  k_qkv   <<<dim3(HD / 16, (BB * SS) / 16, 3), 32, 0, stream>>>(Xb, WT, q, k, vT);
  k_scores<<<dim3(BB * SS), 256, 0, stream>>>(q, k, rpe, station, attnb);
  k_pv    <<<dim3(DD / 16, SS / 16, BB * HH), 32, 0, stream>>>(attnb, vT, atto);
  k_fc    <<<dim3(DMD / 16, (BB * SS) / 16), 32, 0, stream>>>(atto, WT + (size_t)3 * HD * DMD,
                                                              fc_b, hidden, ctx);
  k_ln    <<<dim3(BB * SS), 32, 0, stream>>>(ctx, ln_w, ln_b, out);
}
